// WideNet_36782099923442
// MI455X (gfx1250) — matrix operand order repaired and checked by reading, not compiled, not run on hardware
//
#include <hip/hip_runtime.h>
#include <hip/hip_bf16.h>
#include <cstdint>
#include <type_traits>

#define D_   2048
#define L_   4
#define NE_  16
#define H_   16
#define DH_  128
#define B_   2
#define S_   1024
#define T_   (B_*S_)
#define CAP_ 153
#define CAPR_ 160

typedef __bf16 bf16_t;
typedef __attribute__((ext_vector_type(16))) __bf16 v16bf;
typedef __attribute__((ext_vector_type(8)))  __bf16 v8bf;
typedef __attribute__((ext_vector_type(8)))  float  v8f;
typedef __attribute__((ext_vector_type(4)))  float  v4f;
typedef __attribute__((ext_vector_type(4)))  unsigned int v4u;
typedef __attribute__((ext_vector_type(8)))  int v8i;
typedef __attribute__((ext_vector_type(4)))  int v4i;

// ---------------------------------------------------------------------------
// TDM: issue a 2D tile DMA  global(bf16)[tile 128 rows x 32 cols] -> LDS,
// with hardware padding producing rows of 32 halfs + 8 halfs pad (stride 40),
// and hardware zero-fill for rows/cols beyond (rowsRem, kRem).
// D# fields per CDNA5 ISA 8.3/8.4 (group0: count|lds|gaddr|type=2,
// group1: data_size=2B, pad_enable, pad_interval=16DW, pad_amount=4DW).
// This toolchain's builtin takes 6 args (g0 v4u, g1 v8i, g2 v4i, g3 v4i,
// extra v8i, cpol) — groups 2/3/extra zero for a 2D non-cluster tile.
// ---------------------------------------------------------------------------
__device__ __forceinline__ void tdm_load_tile(const bf16_t* gsrc, unsigned ldsOff,
                                              int rowsRem, int kRem, int strideElems)
{
    unsigned long long ga = (unsigned long long)(const void*)gsrc;
    v4u g0;
    g0[0] = 1u;                                    // count=1, no gather
    g0[1] = ldsOff;                                // lds_addr
    g0[2] = (unsigned)ga;                          // global_addr[31:0]
    g0[3] = (unsigned)((ga >> 32) & 0x01ffffffu)   // global_addr[56:32]
          | (2u << 30);                            // type=2 ("image")
    unsigned td0 = (unsigned)(kRem    < 0 ? 0 : kRem);     // tensor_dim0 (elems)
    unsigned td1 = (unsigned)(rowsRem < 0 ? 0 : rowsRem);  // tensor_dim1 (rows)
    v8i g1;
    g1[0] = (int)((1u << 16)        // data_size = 2 bytes
                | (1u << 20)        // pad_enable
                | (3u << 22)        // pad_interval: 16 DWORDs (one 32-half row)
                | (3u << 25));      // pad_amount:   4 DWORDs (8 halfs)
    g1[1] = (int)((td0 & 0xffffu) << 16);                          // td0 lo16
    g1[2] = (int)(((td0 >> 16) & 0xffffu) | ((td1 & 0xffffu) << 16));
    g1[3] = (int)(((td1 >> 16) & 0xffffu) | (32u << 16));          // tile_dim0=32
    g1[4] = (int)128;                                              // tile_dim1=128
    g1[5] = (int)(unsigned)strideElems;                            // dim0_stride lo
    g1[6] = 0;
    g1[7] = 0;
    v4i gz; gz[0] = 0; gz[1] = 0; gz[2] = 0; gz[3] = 0;
    v8i gz8;
    for (int t = 0; t < 8; ++t) gz8[t] = 0;
    __builtin_amdgcn_tensor_load_to_lds(g0, g1, gz, gz, gz8, 0);
}

__device__ __forceinline__ unsigned lds_offset(const void* p)
{
    return (unsigned)(unsigned long long)p;   // low 32 bits of flat LDS addr = offset
}

// ---------------------------------------------------------------------------
// WMMA-bf16 GEMM:  C[M,N] = A[M,K](bf16) @ W[N,K](bf16)^T + bias[N] (+resid)
// 128x128 block tile, 8 waves (2x4), wave tile 64x32 = 4x2 WMMA tiles.
// K chunks of 32 staged by double-buffered TDM DMA (overlapped with WMMA).
// ---------------------------------------------------------------------------
template <bool OUT_BF16>
__global__ __launch_bounds__(256)
void gemm_bf16_kernel(const bf16_t* __restrict__ A, const bf16_t* __restrict__ W,
                      const float* __restrict__ bias, const float* __restrict__ resid,
                      void* __restrict__ Cout,
                      int M, int N, int K,
                      long sA, long sW, long sBias, long sC, long sRes)
{
    __shared__ __align__(16) bf16_t lds[2][2][128 * 40];   // [buf][A/W][m][k] pad-40

    const int bz = blockIdx.z;
    A += (long)bz * sA;
    W += (long)bz * sW;
    bias += (long)bz * sBias;
    using OutT = typename std::conditional<OUT_BF16, bf16_t, float>::type;
    OutT* C = (OutT*)Cout + (long)bz * sC;
    if (resid) resid += (long)bz * sRes;

    const int n0 = blockIdx.x * 128;
    const int m0 = blockIdx.y * 128;
    const int tid  = threadIdx.x;
    const int lane = tid & 31;
    const int wave = tid >> 5;
    const int wm = wave >> 2;   // 0..1 -> 64-row slab
    const int wn = wave & 3;    // 0..3 -> 32-col slab

    v8f acc[4][2];
    for (int i = 0; i < 4; ++i)
        for (int j = 0; j < 2; ++j)
            for (int r = 0; r < 8; ++r) acc[i][j][r] = 0.0f;

    const bf16_t* Abase = A + (long)m0 * K;
    const bf16_t* Wbase = W + (long)n0 * K;
    unsigned offA[2] = { lds_offset(&lds[0][0][0]), lds_offset(&lds[1][0][0]) };
    unsigned offB[2] = { lds_offset(&lds[0][1][0]), lds_offset(&lds[1][1][0]) };

    const bool issuer = (wave == 0);          // wave-uniform
    if (issuer) {
        tdm_load_tile(Abase, offA[0], M - m0, K, K);
        tdm_load_tile(Wbase, offB[0], N - n0, K, K);
    }

    const int nIter = K / 32;
    for (int it = 0; it < nIter; ++it) {
        const int cur = it & 1;
        __syncthreads();   // all waves done reading buf[1-cur] from iteration it-1
        if (issuer) {
            if (it + 1 < nIter) {
                const int k1 = (it + 1) * 32;
                tdm_load_tile(Abase + k1, offA[1 - cur], M - m0, K - k1, K);
                tdm_load_tile(Wbase + k1, offB[1 - cur], N - n0, K - k1, K);
                __builtin_amdgcn_s_wait_tensorcnt(2);  // in-order: chunk `it` done
            } else {
                __builtin_amdgcn_s_wait_tensorcnt(0);
            }
        }
        __syncthreads();   // chunk `it` visible to all waves

        const bf16_t* lA = &lds[cur][0][0];
        const bf16_t* lB = &lds[cur][1][0];

        // A-fragment (16x32): lane<16 -> K runs [0,8)+[16,24); lane>=16 -> [8,16)+[24,32)
        v16bf aF[4];
        const int koffA = (lane < 16) ? 0 : 8;
        for (int i = 0; i < 4; ++i) {
            const bf16_t* p = &lA[(wm * 64 + i * 16 + (lane & 15)) * 40];
            v8bf lo = *(const v8bf*)(p + koffA);
            v8bf hi = *(const v8bf*)(p + koffA + 16);
            v16bf f;
            for (int t = 0; t < 8; ++t) { f[t] = lo[t]; f[8 + t] = hi[t]; }
            aF[i] = f;
        }
        // B-fragment (32x16): lane = column n, lane<16 -> K [0,8)+[16,24); lane>=16 -> [8,16)+[24,32)
        v16bf bF[2];
        const int koffB = (lane < 16) ? 0 : 8;
        for (int j = 0; j < 2; ++j) {
            const bf16_t* p = &lB[(wn * 32 + j * 16 + (lane & 15)) * 40 + koffB];
            v8bf b0 = *(const v8bf*)(p);
            v8bf b1 = *(const v8bf*)(p + 16);
            v16bf f;
            for (int t = 0; t < 8; ++t) { f[t] = b0[t]; f[8 + t] = b1[t]; }
            bF[j] = f;
        }

        for (int i = 0; i < 4; ++i)
            for (int j = 0; j < 2; ++j)
                acc[i][j] = __builtin_amdgcn_wmma_f32_16x16x32_bf16(
                    false, aF[i], false, bF[j], (short)0, acc[i][j], false, false);
    }

    // epilogue: C layout = lane -> N col, VGPR r -> row r (+8 for lanes 16..31)
    const int rbase = (lane < 16) ? 0 : 8;
    for (int i = 0; i < 4; ++i) {
        for (int j = 0; j < 2; ++j) {
            int col = n0 + wn * 32 + j * 16 + (lane & 15);
            float bv = bias[col];
            for (int r = 0; r < 8; ++r) {
                int row = m0 + wm * 64 + i * 16 + rbase + r;
                if (row < M) {
                    long idx = (long)row * N + col;
                    float val = acc[i][j][r] + bv;
                    if (!OUT_BF16 && resid) val += resid[idx];
                    C[idx] = (OutT)val;
                }
            }
        }
    }
}

// ---------------------------------------------------------------------------
// Flash attention, 1 wave per (b, head, 16-row q tile). bf16 qkv in, bf16 o out.
// ---------------------------------------------------------------------------
__global__ __launch_bounds__(32)
void attn_kernel(const bf16_t* __restrict__ qkv, bf16_t* __restrict__ o)
{
    __shared__ __align__(16) bf16_t lP[16 * 40];

    const int lane = threadIdx.x & 31;
    const int qt = blockIdx.x;
    const int hh = blockIdx.y;
    const int bb = blockIdx.z;
    const long rs = 3L * D_;

    const bf16_t* qb = qkv + (long)(bb * S_) * rs + hh * DH_;
    const bf16_t* kb = qb + D_;
    const bf16_t* vb = qb + 2 * D_;

    v16bf aQ[4];
    {
        int m = qt * 16 + (lane & 15);
        const bf16_t* qrow = qb + (long)m * rs;
        int ko = (lane < 16) ? 0 : 8;
        for (int kd = 0; kd < 4; ++kd) {
            const bf16_t* p = qrow + kd * 32 + ko;
            v8bf lo = *(const v8bf*)(p);
            v8bf hi = *(const v8bf*)(p + 16);
            v16bf f;
            for (int t = 0; t < 8; ++t) { f[t] = lo[t]; f[8 + t] = hi[t]; }
            aQ[kd] = f;
        }
    }

    v8f accO[8];
    for (int j = 0; j < 8; ++j)
        for (int r = 0; r < 8; ++r) accO[j][r] = 0.0f;
    float rowM[8], rowS[8];
    for (int r = 0; r < 8; ++r) { rowM[r] = -1e30f; rowS[r] = 0.0f; }

    const int rbase = (lane < 16) ? 0 : 8;
    const float sc = 0.08838834764831845f;   // 1/sqrt(128)

    for (int kb0 = 0; kb0 < S_; kb0 += 32) {
        v8f c0, c1;
        for (int r = 0; r < 8; ++r) { c0[r] = 0.0f; c1[r] = 0.0f; }
        for (int kd = 0; kd < 4; ++kd) {
            int ko = kd * 32 + ((lane < 16) ? 0 : 8);
            for (int nt = 0; nt < 2; ++nt) {
                int kvc = kb0 + nt * 16 + (lane & 15);
                const bf16_t* kr = kb + (long)kvc * rs + ko;
                v8bf a0 = *(const v8bf*)(kr);
                v8bf a1 = *(const v8bf*)(kr + 16);
                v16bf f;
                for (int t = 0; t < 8; ++t) { f[t] = a0[t]; f[8 + t] = a1[t]; }
                if (nt == 0)
                    c0 = __builtin_amdgcn_wmma_f32_16x16x32_bf16(false, aQ[kd], false, f,
                                                                 (short)0, c0, false, false);
                else
                    c1 = __builtin_amdgcn_wmma_f32_16x16x32_bf16(false, aQ[kd], false, f,
                                                                 (short)0, c1, false, false);
            }
        }

        float alpha[8];
        for (int r = 0; r < 8; ++r) {
            float s0 = c0[r] * sc, s1 = c1[r] * sc;
            float mx = fmaxf(s0, s1);
            for (int m = 1; m < 16; m <<= 1) mx = fmaxf(mx, __shfl_xor(mx, m, 32));
            float mnew = fmaxf(rowM[r], mx);
            float p0 = __expf(s0 - mnew), p1 = __expf(s1 - mnew);
            float ps = p0 + p1;
            for (int m = 1; m < 16; m <<= 1) ps += __shfl_xor(ps, m, 32);
            float a = __expf(rowM[r] - mnew);
            rowS[r] = rowS[r] * a + ps;
            rowM[r] = mnew;
            alpha[r] = a;
            c0[r] = p0; c1[r] = p1;
        }
        for (int j = 0; j < 8; ++j)
            for (int r = 0; r < 8; ++r) accO[j][r] *= alpha[r];

        __syncthreads();
        for (int r = 0; r < 8; ++r) {
            lP[(rbase + r) * 40 + (lane & 15)]      = (bf16_t)c0[r];
            lP[(rbase + r) * 40 + 16 + (lane & 15)] = (bf16_t)c1[r];
        }
        __syncthreads();
        v16bf aP;
        {
            const bf16_t* p = &lP[(lane & 15) * 40 + ((lane < 16) ? 0 : 16)];
            v8bf lo = *(const v8bf*)(p);
            v8bf hi = *(const v8bf*)(p + 8);
            for (int t = 0; t < 8; ++t) { aP[t] = lo[t]; aP[8 + t] = hi[t]; }
        }

        int kvr = kb0 + ((lane < 16) ? 0 : 16);
        for (int j = 0; j < 8; ++j) {
            int dcol = j * 16 + (lane & 15);
            const bf16_t* vcol = vb + (long)kvr * rs + dcol;
            v16bf f;
            for (int t = 0; t < 16; ++t) f[t] = vcol[(long)t * rs];
            accO[j] = __builtin_amdgcn_wmma_f32_16x16x32_bf16(false, aP, false, f,
                                                              (short)0, accO[j], false, false);
        }
    }

    for (int j = 0; j < 8; ++j) {
        int dcol = j * 16 + (lane & 15);
        for (int r = 0; r < 8; ++r) {
            int m = qt * 16 + rbase + r;
            o[(long)(bb * S_ + m) * D_ + hh * DH_ + dcol] = (bf16_t)(accO[j][r] / rowS[r]);
        }
    }
}

// ---------------------------------------------------------------------------
__global__ __launch_bounds__(256)
void ln_kernel(const float* __restrict__ x, const float* __restrict__ g,
               const float* __restrict__ be, float* __restrict__ out,
               bf16_t* __restrict__ outb)
{
    __shared__ float s1[256], s2[256];
    const int t = blockIdx.x;
    const float* xr = x + (long)t * D_;
    float s = 0.f, ss = 0.f;
    for (int d = threadIdx.x; d < D_; d += 256) { float v = xr[d]; s += v; ss += v * v; }
    s1[threadIdx.x] = s; s2[threadIdx.x] = ss;
    __syncthreads();
    for (int o2 = 128; o2 > 0; o2 >>= 1) {
        if (threadIdx.x < o2) { s1[threadIdx.x] += s1[threadIdx.x + o2];
                                s2[threadIdx.x] += s2[threadIdx.x + o2]; }
        __syncthreads();
    }
    float mu = s1[0] / (float)D_;
    float var = s2[0] / (float)D_ - mu * mu;
    float rstd = rsqrtf(var + 1e-5f);
    for (int d = threadIdx.x; d < D_; d += 256) {
        float v = (xr[d] - mu) * rstd * g[d] + be[d];
        out[(long)t * D_ + d] = v;
        outb[(long)t * D_ + d] = (bf16_t)v;
    }
}

__global__ __launch_bounds__(256)
void gate_kernel(const float* __restrict__ h, const float* __restrict__ wg,
                 int* __restrict__ top, float* __restrict__ pval,
                 float* __restrict__ ohSum, float* __restrict__ probsSum)
{
    __shared__ float acc[NE_];
    const int t = blockIdx.x;
    if (threadIdx.x < NE_) acc[threadIdx.x] = 0.f;
    __syncthreads();
    float loc[NE_];
    for (int e = 0; e < NE_; ++e) loc[e] = 0.f;
    const float* hr = h + (long)t * D_;
    for (int d = threadIdx.x; d < D_; d += 256) {
        float xv = hr[d];
        const float* wr = wg + (long)d * NE_;
        for (int e = 0; e < NE_; ++e) loc[e] += xv * wr[e];
    }
    for (int e = 0; e < NE_; ++e) atomicAdd(&acc[e], loc[e]);
    __syncthreads();
    if (threadIdx.x == 0) {
        float l[NE_]; float mx = -1e30f; int am = 0;
        for (int e = 0; e < NE_; ++e) { l[e] = acc[e]; if (l[e] > mx) { mx = l[e]; am = e; } }
        float sum = 0.f;
        for (int e = 0; e < NE_; ++e) { l[e] = __expf(l[e] - mx); sum += l[e]; }
        float inv = 1.f / sum;
        for (int e = 0; e < NE_; ++e) atomicAdd(&probsSum[e], l[e] * inv);
        top[t] = am;
        pval[t] = l[am] * inv;
        atomicAdd(&ohSum[am], 1.0f);
    }
}

__global__ void route_kernel(const int* __restrict__ top, int* __restrict__ keep,
                             int* __restrict__ slot, const float* __restrict__ ohSum,
                             const float* __restrict__ probsSum, float* __restrict__ auxAcc)
{
    int cnt[NE_];
    for (int e = 0; e < NE_; ++e) cnt[e] = 0;
    for (int t = 0; t < T_; ++t) {
        int e = top[t];
        int r = cnt[e]++;
        keep[t] = (r < CAP_) ? 1 : 0;
        slot[t] = e * CAPR_ + ((r < CAP_) ? r : 0);
    }
    float a = 0.f;
    for (int e = 0; e < NE_; ++e) a += ohSum[e] * probsSum[e];
    *auxAcc += (float)NE_ * a / ((float)T_ * (float)T_);
}

__global__ void zero_sums_kernel(float* ohSum, float* probsSum)
{
    if (threadIdx.x < NE_) { ohSum[threadIdx.x] = 0.f; probsSum[threadIdx.x] = 0.f; }
}
__global__ void zero_aux_kernel(float* auxAcc) { *auxAcc = 0.f; }

__global__ __launch_bounds__(256)
void gather_kernel(const bf16_t* __restrict__ hb, const int* __restrict__ keep,
                   const int* __restrict__ slot, bf16_t* __restrict__ Xeb)
{
    const int t = blockIdx.y;
    if (!keep[t]) return;
    const int d = blockIdx.x * 256 + threadIdx.x;
    Xeb[(long)slot[t] * D_ + d] = hb[(long)t * D_ + d];
}

__global__ __launch_bounds__(256)
void scatter_kernel(const float* __restrict__ Ye, const int* __restrict__ keep,
                    const int* __restrict__ slot, const float* __restrict__ pval,
                    float* __restrict__ x)
{
    const int t = blockIdx.y;
    if (!keep[t]) return;
    const int d = blockIdx.x * 256 + threadIdx.x;
    x[(long)t * D_ + d] += pval[t] * Ye[(long)slot[t] * D_ + d];
}

__global__ __launch_bounds__(256)
void embed_kernel(const int* __restrict__ tokens, const float* __restrict__ emb,
                  float* __restrict__ x)
{
    const int t = blockIdx.y;
    const int d = blockIdx.x * 256 + threadIdx.x;
    x[(long)t * D_ + d] = emb[(long)tokens[t] * D_ + d];
}

__global__ __launch_bounds__(256)
void cvt_bf16_kernel(const float* __restrict__ in, bf16_t* __restrict__ out, long n)
{
    long i = ((long)blockIdx.x * 256 + threadIdx.x) * 4;
    if (i + 3 < n) {
        v4f v = *(const v4f*)(in + i);
        out[i]     = (bf16_t)v[0];
        out[i + 1] = (bf16_t)v[1];
        out[i + 2] = (bf16_t)v[2];
        out[i + 3] = (bf16_t)v[3];
    }
}

__global__ __launch_bounds__(256)
void finalize_kernel(const float* __restrict__ x, const float* __restrict__ auxAcc,
                     float* __restrict__ out)
{
    const long i = (long)blockIdx.x * 256 + threadIdx.x;
    if (i < (long)T_ * D_) out[i] = x[i];
    if (i == 0) out[(long)T_ * D_] = *auxAcc;
}

// ---------------------------------------------------------------------------
extern "C" void kernel_launch(void* const* d_in, const int* in_sizes, int n_in,
                              void* d_out, int out_size, void* d_ws, size_t ws_size,
                              hipStream_t stream)
{
    (void)in_sizes; (void)n_in; (void)out_size; (void)ws_size;

    const int*   tokens = (const int*)d_in[0];
    const float* emb    = (const float*)d_in[1];
    const float* w_qkv  = (const float*)d_in[2];
    const float* b_qkv  = (const float*)d_in[3];
    const float* w_out  = (const float*)d_in[4];
    const float* b_out  = (const float*)d_in[5];
    const float* attn_g = (const float*)d_in[6];
    const float* attn_b = (const float*)d_in[7];
    const float* moe_g  = (const float*)d_in[8];
    const float* moe_b  = (const float*)d_in[9];
    const float* w_gate = (const float*)d_in[10];
    const float* w_exp  = (const float*)d_in[11];
    const float* b_exp  = (const float*)d_in[12];
    float* out = (float*)d_out;

    const long TD = (long)T_ * D_;
    float* ws = (float*)d_ws;
    float* x        = ws;                                // T*D f32
    float* h        = x + TD;                            // T*D f32
    float* Ye       = h + TD;                            // NE*CAPR*D f32
    float* pval     = Ye + (long)NE_ * CAPR_ * D_;       // T
    float* ohSum    = pval + T_;                         // NE
    float* probsSum = ohSum + NE_;                       // NE
    float* auxAcc   = probsSum + NE_;                    // 1
    int* top  = (int*)(auxAcc + 1);                      // T
    int* keep = top + T_;                                // T
    int* slot = keep + T_;                               // T
    // bf16 region, 64B aligned
    bf16_t* hb    = (bf16_t*)(((size_t)(slot + T_) + 63) & ~(size_t)63);
    bf16_t* ob    = hb + TD;                             // T*D
    bf16_t* qkvb  = ob + TD;                             // T*3D
    bf16_t* Xeb   = qkvb + 3 * TD;                       // NE*CAPR*D
    bf16_t* wqkvb = Xeb + (long)NE_ * CAPR_ * D_;        // 3D*D
    bf16_t* woutb = wqkvb + 3L * D_ * D_;                // D*D
    bf16_t* wexpb = woutb + (long)D_ * D_;               // NE*D*D

    zero_aux_kernel<<<1, 1, 0, stream>>>(auxAcc);
    embed_kernel<<<dim3(D_ / 256, T_), 256, 0, stream>>>(tokens, emb, x);

    // one-time per launch weight conversion to bf16
    cvt_bf16_kernel<<<(3L * D_ * D_) / 1024, 256, 0, stream>>>(w_qkv, wqkvb, 3L * D_ * D_);
    cvt_bf16_kernel<<<((long)D_ * D_) / 1024, 256, 0, stream>>>(w_out, woutb, (long)D_ * D_);
    cvt_bf16_kernel<<<((long)NE_ * D_ * D_) / 1024, 256, 0, stream>>>(w_exp, wexpb,
                                                                      (long)NE_ * D_ * D_);

    for (int i = 0; i < L_; ++i) {
        // ---- attention block ----
        ln_kernel<<<T_, 256, 0, stream>>>(x, attn_g + (long)i * D_, attn_b + (long)i * D_,
                                          h, hb);
        gemm_bf16_kernel<true><<<dim3(3 * D_ / 128, T_ / 128, 1), 256, 0, stream>>>(
            hb, wqkvb, b_qkv, nullptr, qkvb, T_, 3 * D_, D_, 0, 0, 0, 0, 0);
        attn_kernel<<<dim3(S_ / 16, H_, B_), 32, 0, stream>>>(qkvb, ob);
        gemm_bf16_kernel<false><<<dim3(D_ / 128, T_ / 128, 1), 256, 0, stream>>>(
            ob, woutb, b_out, x, x, T_, D_, D_, 0, 0, 0, 0, 0);

        // ---- MoE block ----
        ln_kernel<<<T_, 256, 0, stream>>>(x, moe_g + (long)i * D_, moe_b + (long)i * D_,
                                          h, hb);
        zero_sums_kernel<<<1, 32, 0, stream>>>(ohSum, probsSum);
        gate_kernel<<<T_, 256, 0, stream>>>(h, w_gate, top, pval, ohSum, probsSum);
        route_kernel<<<1, 1, 0, stream>>>(top, keep, slot, ohSum, probsSum, auxAcc);
        gather_kernel<<<dim3(D_ / 256, T_), 256, 0, stream>>>(hb, keep, slot, Xeb);
        gemm_bf16_kernel<false><<<dim3(D_ / 128, (CAPR_ + 127) / 128, NE_), 256, 0, stream>>>(
            Xeb, wexpb, b_exp, nullptr, Ye, CAPR_, D_, D_,
            (long)CAPR_ * D_, (long)D_ * D_, (long)D_, (long)CAPR_ * D_, 0);
        scatter_kernel<<<dim3(D_ / 256, T_), 256, 0, stream>>>(Ye, keep, slot, pval, x);
    }

    finalize_kernel<<<(TD + 255) / 256, 256, 0, stream>>>(x, auxAcc, out);
}
